// GCNEncoder_15530601742663
// MI455X (gfx1250) — matrix operand order repaired and checked by reading, not compiled, not run on hardware
//
#include <hip/hip_runtime.h>
#include <hip/hip_bf16.h>

#define B_N 16
#define C_N 64
#define F_N 128
#define TP_N 100
#define E_N 576
#define G_TOT (B_N * TP_N)       // 1600 graphs
#define N_TOT (G_TOT * C_N)      // 102400 nodes
#define K_KEEP 58
#define NK_TOT (G_TOT * K_KEEP)  // 92800 kept nodes
#define BN_EPS 1e-5f

typedef __attribute__((ext_vector_type(16))) _Float16 v16h;
typedef __attribute__((ext_vector_type(8)))  _Float16 v8h;
typedef __attribute__((ext_vector_type(8)))  float    v8f;

union V16H { v16h v; _Float16 e[16]; };
union V8H  { v8h  v; _Float16 e[8];  };
union V8F  { v8f  v; float    e[8];  };

#define WMMA_F16(A, B, C) \
  __builtin_amdgcn_wmma_f32_16x16x32_f16(false, (A), false, (B), (short)0, (C), false, false)

// ---------------- fragment-major layouts (ISA 7.12.2, wave32) ----------------
// A-fragment (16x32 f16 tile): lane = 16*half + (m&15); lane's 16 elems contiguous.
//   element idx for in-tile k': hi=k'>>4, rem=k'&15, half=rem>>3, idx=hi*8+(rem&7)
__device__ inline int a_slot(int m, int k, int KT) {
  int mt = m >> 4, kt = k >> 5, kk = k & 31;
  int hi = kk >> 4, rem = kk & 15;
  int half = rem >> 3, idx = hi * 8 + (rem & 7);
  int lane = half * 16 + (m & 15);
  return ((mt * KT + kt) * 32 + lane) * 16 + idx;
}
// B-fragment (32x16 f16 tile): lane = 16*((k'&15)>>3) + (n&15); idx = (k'>>4)*8 + (k'&7)
__device__ inline int b_slot(int k, int n, int TC) {
  int kt = k >> 5, kk = k & 31, tc = n >> 4;
  int half = (kk & 15) >> 3, idx = (kk >> 4) * 8 + (kk & 7);
  int lane = half * 16 + (n & 15);
  return ((kt * TC + tc) * 32 + lane) * 16 + idx;
}
// load one lane's 32-byte fragment (2x b128)
__device__ inline v16h frag_ld(const _Float16* base, int tile, int lane) {
  return *(const v16h*)(base + ((size_t)tile * 32 + lane) * 16);
}

// ================= K0: adjacency + weight swizzle (runs once) =================
__global__ __launch_bounds__(256) void k0_prep(const int* __restrict__ ei,
                                               const float* __restrict__ ew,
                                               const float* __restrict__ W1,
                                               const float* __restrict__ W2,
                                               _Float16* W1f, _Float16* W2f,
                                               _Float16* A1f_g, float* Apf_g) {
  __shared__ float A1f[64][64];
  __shared__ float Apf[64][64];
  __shared__ float deg[64], degp[64], dis[64], disp[64];
  int tid = threadIdx.x;
  // swizzle W1 (128x128) into B-layout [ks=4][tc=8][32][16]
  for (int e = tid; e < 128 * 128; e += 256)
    W1f[b_slot(e >> 7, e & 127, 8)] = (_Float16)W1[e];
  // swizzle W2 (128x64) into B-layout [ks=4][tc=4][32][16]
  for (int e = tid; e < 128 * 64; e += 256)
    W2f[b_slot(e >> 6, e & 63, 4)] = (_Float16)W2[e];
  for (int e = tid; e < 4096; e += 256) { A1f[e >> 6][e & 63] = 0.f; Apf[e >> 6][e & 63] = 0.f; }
  if (tid < 64) { deg[tid] = 1.f; degp[tid] = 1.f; }  // self-loop weight 1
  __syncthreads();
  if (tid == 0) {
    for (int e = 0; e < E_N; ++e) {
      int c = ei[E_N + e];
      deg[c]  += ew[e];
      degp[c] += 1.f;
    }
  }
  __syncthreads();
  if (tid < 64) { dis[tid] = rsqrtf(deg[tid]); disp[tid] = rsqrtf(degp[tid]); }
  __syncthreads();
  if (tid == 0) {
    for (int e = 0; e < E_N; ++e) {
      int r = ei[e], c = ei[E_N + e];
      A1f[c][r] += dis[r] * ew[e] * dis[c];
      Apf[c][r] += disp[r] * disp[c];
    }
    for (int i = 0; i < 64; ++i) {
      A1f[i][i] += dis[i] * dis[i];
      Apf[i][i] += disp[i] * disp[i];
    }
  }
  __syncthreads();
  // A1 (64x64) into A-layout [tr=4][ks=2][32][16]; Ap stays row-major f32
  for (int e = tid; e < 4096; e += 256) {
    A1f_g[a_slot(e >> 6, e & 63, 2)] = (_Float16)A1f[e >> 6][e & 63];
    Apf_g[e] = Apf[e >> 6][e & 63];
  }
}

// ================= K1: per-graph fused  gc1 = A1 @ (X_g @ W1) + b1 =================
__global__ __launch_bounds__(256) void k1_gc1(const float* __restrict__ h,
                                              const _Float16* __restrict__ W1f,
                                              const _Float16* __restrict__ A1f,
                                              const float* __restrict__ b1,
                                              float* __restrict__ gc1_raw) {
  __shared__ _Float16 Xs[8192]; // A-layout [tr=4][kt=4][32][16]
  __shared__ _Float16 Hs[8192]; // B-layout [kt=2][tc=8][32][16]
  int g = blockIdx.x, tid = threadIdx.x;
  int b = g / TP_N, tp = g % TP_N;
  const float* hg = h + (size_t)b * 64 * 128 * TP_N + tp;
  // gather X_g[m][k] = h[b][m][k][tp], packed straight into A-fragment layout
#pragma unroll
  for (int c = 0; c < 2; ++c) {
    int s0 = tid * 32 + c * 16;
    int tile = s0 >> 9, lane = (s0 >> 4) & 31;
    int tr = tile >> 2, kt = tile & 3;
    int m = tr * 16 + (lane & 15), half = lane >> 4;
    V16H tmp;
#pragma unroll
    for (int idx = 0; idx < 16; ++idx) {
      int k = kt * 32 + (idx >> 3) * 16 + half * 8 + (idx & 7);
      tmp.e[idx] = (_Float16)hg[((size_t)m * 128 + k) * TP_N];
    }
    *(v16h*)(Xs + s0) = tmp.v;   // one 32B LDS store
  }
  __syncthreads();

  int lane = tid & 31, wave = tid >> 5, half = lane >> 4, l16 = lane & 15;
  int trow = wave >> 1;                     // fixed per wave -> A-frag reuse
  V8F acc[4];
#pragma unroll
  for (int q = 0; q < 4; ++q) acc[q].v = (v8f){0, 0, 0, 0, 0, 0, 0, 0};

  // GEMM1: H(64x128) = X(64x128) @ W1(128x128)
#pragma unroll
  for (int ks = 0; ks < 4; ++ks) {
    v16h a = frag_ld(Xs, trow * 4 + ks, lane);          // LDS b128 x2
#pragma unroll
    for (int q = 0; q < 4; ++q) {
      int tcol = (wave & 1) * 4 + q;
      v16h bb = frag_ld(W1f, ks * 8 + tcol, lane);      // global b128 x2
      acc[q].v = WMMA_F16(a, bb, acc[q].v);
    }
  }
  // D -> B-fragment layout of Hs: 8 contiguous halfwords per tile per lane
#pragma unroll
  for (int q = 0; q < 4; ++q) {
    int tcol = (wave & 1) * 4 + q;
    V8H pk;
#pragma unroll
    for (int i = 0; i < 8; ++i) pk.e[i] = (_Float16)acc[q].e[i];
    *(v8h*)(Hs + ((((trow >> 1) * 8 + tcol) * 32 + (half * 16 + l16)) * 16 + (trow & 1) * 8)) = pk.v;
    acc[q].v = (v8f){0, 0, 0, 0, 0, 0, 0, 0};
  }
  __syncthreads();
  // GEMM2: out(64x128) = A1(64x64) @ H(64x128)
#pragma unroll
  for (int ks = 0; ks < 2; ++ks) {
    v16h a = frag_ld(A1f, trow * 2 + ks, lane);         // global b128 x2
#pragma unroll
    for (int q = 0; q < 4; ++q) {
      int tcol = (wave & 1) * 4 + q;
      v16h bb = frag_ld(Hs, ks * 8 + tcol, lane);       // LDS b128 x2
      acc[q].v = WMMA_F16(a, bb, acc[q].v);
    }
  }
#pragma unroll
  for (int q = 0; q < 4; ++q) {
    int tcol = (wave & 1) * 4 + q;
#pragma unroll
    for (int i = 0; i < 8; ++i) {
      int m = trow * 16 + half * 8 + i;
      int f = tcol * 16 + l16;
      gc1_raw[((size_t)g * 64 + m) * 128 + f] = acc[q].e[i] + b1[f];
    }
  }
}

// ================= per-graph BN partial sums =================
__global__ __launch_bounds__(128) void k1b_stats(const float* __restrict__ gc1_raw,
                                                 float* psum, float* psumsq) {
  int g = blockIdx.x, f = threadIdx.x;
  float s = 0.f, q = 0.f;
  for (int m = 0; m < 64; ++m) {
    float v = gc1_raw[((size_t)g * 64 + m) * 128 + f];
    s += v; q += v * v;
  }
  psum[g * 128 + f] = s; psumsq[g * 128 + f] = q;
}

__global__ __launch_bounds__(64) void k3b_stats(const float* __restrict__ gc2_raw,
                                                float* psum, float* psumsq) {
  int g = blockIdx.x, f = threadIdx.x;
  float s = 0.f, q = 0.f;
  for (int p = 0; p < K_KEEP; ++p) {
    float v = gc2_raw[(size_t)g * 4096 + p * 64 + f];
    s += v; q += v * v;
  }
  psum[g * 64 + f] = s; psumsq[g * 64 + f] = q;
}

// ================= BN reduction (deterministic, fixed order) =================
__global__ void k_bnred(const float* __restrict__ psum, const float* __restrict__ psumsq,
                        const float* __restrict__ gam, const float* __restrict__ bet,
                        float* scale, float* shift, int nf, int ng, float invN) {
  int f = threadIdx.x;
  if (f >= nf) return;
  float s = 0.f, q = 0.f;
  for (int g = 0; g < ng; ++g) { s += psum[g * nf + f]; q += psumsq[g * nf + f]; }
  float mean = s * invN;
  float var  = q * invN - mean * mean;
  float sc   = gam[f] * rsqrtf(var + BN_EPS);
  scale[f] = sc;
  shift[f] = bet[f] - mean * sc;
}

// ====== K2: BN1+PReLU, score, top-k, xk & A2 emitted in fragment layout ======
__global__ __launch_bounds__(256) void k2_pool(const float* __restrict__ gc1_raw,
                                               const float* __restrict__ scale1,
                                               const float* __restrict__ shift1,
                                               const float* __restrict__ a1p,
                                               const float* __restrict__ Wp,
                                               const float* __restrict__ bpp,
                                               const float* __restrict__ Apf,
                                               const int* __restrict__ ei,
                                               const float* __restrict__ ew,
                                               _Float16* __restrict__ xk_f,
                                               _Float16* __restrict__ A2f_h) {
  __shared__ _Float16 xs[64][128];
  __shared__ float A2f[64][64];
  __shared__ float tvec[64], svec[64], dis2[64];
  __shared__ int rank[64], node_of[64];
  int g = blockIdx.x, tid = threadIdx.x;
  float a1 = a1p[0];
  for (int e = tid; e < 8192; e += 256) {
    int m = e >> 7, f = e & 127;
    float v = gc1_raw[((size_t)g * 64 + m) * 128 + f] * scale1[f] + shift1[f];
    xs[m][f] = (_Float16)(v >= 0.f ? v : a1 * v);
  }
  __syncthreads();
  if (tid < 64) {                                   // t = x @ Wp
    float acc = 0.f;
    for (int f = 0; f < 128; ++f) acc += (float)xs[tid][f] * Wp[f];
    tvec[tid] = acc;
  }
  __syncthreads();
  if (tid < 64) {                                   // s = tanh(Ap @ t + bp)
    float acc = bpp[0];
    for (int j = 0; j < 64; ++j) acc += Apf[tid * 64 + j] * tvec[j];
    svec[tid] = tanhf(acc);
  }
  __syncthreads();
  if (tid < 64) {                                   // top_k ranks (desc, ties -> lower idx)
    float si = svec[tid];
    int r = 0;
    for (int j = 0; j < 64; ++j) {
      float sj = svec[j];
      r += (sj > si) || (sj == si && j < tid);
    }
    rank[tid] = r;
    if (r < K_KEEP) node_of[r] = tid;
  }
  __syncthreads();
  if (tid == 0) {                                   // pruned-graph degrees (deterministic)
    for (int p = 0; p < K_KEEP; ++p) dis2[p] = 1.f;
    for (int e = 0; e < E_N; ++e) {
      int r = ei[e], c = ei[E_N + e];
      if (rank[r] < K_KEEP && rank[c] < K_KEEP) dis2[rank[c]] += ew[e];
    }
    for (int p = 0; p < K_KEEP; ++p) dis2[p] = rsqrtf(dis2[p]);
  }
  for (int e = tid; e < 4096; e += 256) A2f[e >> 6][e & 63] = 0.f;
  __syncthreads();
  if (tid == 0) {
    for (int e = 0; e < E_N; ++e) {
      int r = ei[e], c = ei[E_N + e];
      int pr = rank[r], pc = rank[c];
      if (pr < K_KEEP && pc < K_KEEP) A2f[pc][pr] += dis2[pr] * ew[e] * dis2[pc];
    }
    for (int p = 0; p < K_KEEP; ++p) A2f[p][p] += dis2[p] * dis2[p];
  }
  __syncthreads();
  // xk (pos-major, score-scaled, zero-padded) -> A-layout [tr=4][kt=4][32][16], coalesced stores
  for (int s = tid; s < 8192; s += 256) {
    int tile = s >> 9, lane = (s >> 4) & 31, idx = s & 15;
    int m = (tile >> 2) * 16 + (lane & 15), half = lane >> 4;
    int k = (tile & 3) * 32 + (idx >> 3) * 16 + half * 8 + (idx & 7);
    float v = 0.f;
    if (m < K_KEEP) { int nd = node_of[m]; v = (float)xs[nd][k] * svec[nd]; }
    xk_f[(size_t)g * 8192 + s] = (_Float16)v;
  }
  // A2 -> A-layout [tr=4][ks=2][32][16], coalesced stores
  for (int s = tid; s < 4096; s += 256) {
    int tile = s >> 9, lane = (s >> 4) & 31, idx = s & 15;
    int m = (tile >> 1) * 16 + (lane & 15), half = lane >> 4;
    int k = (tile & 1) * 32 + (idx >> 3) * 16 + half * 8 + (idx & 7);
    A2f_h[(size_t)g * 4096 + s] = (_Float16)A2f[m][k];
  }
}

// ================= K3: per-graph  gc2 = A2 @ (xk @ W2) + b2 =================
__global__ __launch_bounds__(256) void k3_gc2(const _Float16* __restrict__ xk_f,
                                              const _Float16* __restrict__ W2f,
                                              const _Float16* __restrict__ A2f_h,
                                              const float* __restrict__ b2,
                                              float* __restrict__ gc2_raw) {
  __shared__ _Float16 Hs[4096]; // B-layout [kt=2][tc=4][32][16]
  int g = blockIdx.x, tid = threadIdx.x;
  const _Float16* xg = xk_f + (size_t)g * 8192;
  const _Float16* ag = A2f_h + (size_t)g * 4096;
  int lane = tid & 31, wave = tid >> 5, half = lane >> 4, l16 = lane & 15;
  int trow = wave >> 1;
  V8F acc[2];
#pragma unroll
  for (int q = 0; q < 2; ++q) acc[q].v = (v8f){0, 0, 0, 0, 0, 0, 0, 0};

  // GEMM1: H(64x64) = xk(64x128) @ W2(128x64) — all operands direct from global
#pragma unroll
  for (int ks = 0; ks < 4; ++ks) {
    v16h a = frag_ld(xg, trow * 4 + ks, lane);
#pragma unroll
    for (int q = 0; q < 2; ++q) {
      int tcol = (wave & 1) * 2 + q;
      v16h bb = frag_ld(W2f, ks * 4 + tcol, lane);
      acc[q].v = WMMA_F16(a, bb, acc[q].v);
    }
  }
#pragma unroll
  for (int q = 0; q < 2; ++q) {
    int tcol = (wave & 1) * 2 + q;
    V8H pk;
#pragma unroll
    for (int i = 0; i < 8; ++i) pk.e[i] = (_Float16)acc[q].e[i];
    *(v8h*)(Hs + ((((trow >> 1) * 4 + tcol) * 32 + (half * 16 + l16)) * 16 + (trow & 1) * 8)) = pk.v;
    acc[q].v = (v8f){0, 0, 0, 0, 0, 0, 0, 0};
  }
  __syncthreads();
  // GEMM2: out(64x64) = A2(64x64) @ H(64x64)
#pragma unroll
  for (int ks = 0; ks < 2; ++ks) {
    v16h a = frag_ld(ag, trow * 2 + ks, lane);
#pragma unroll
    for (int q = 0; q < 2; ++q) {
      int tcol = (wave & 1) * 2 + q;
      v16h bb = frag_ld(Hs, ks * 4 + tcol, lane);
      acc[q].v = WMMA_F16(a, bb, acc[q].v);
    }
  }
#pragma unroll
  for (int q = 0; q < 2; ++q) {
    int tcol = (wave & 1) * 2 + q;
#pragma unroll
    for (int i = 0; i < 8; ++i) {
      int m = trow * 16 + half * 8 + i;
      int n = tcol * 16 + l16;
      gc2_raw[(size_t)g * 4096 + m * 64 + n] = acc[q].e[i] + b2[n];
    }
  }
}

// ================= K5: BN2 + PReLU + mean pool, write outputs =================
__global__ __launch_bounds__(64) void k5_final(const float* __restrict__ gc2_raw,
                                               const float* __restrict__ scale2,
                                               const float* __restrict__ shift2,
                                               const float* __restrict__ a2p,
                                               float* __restrict__ out) {
  int g = blockIdx.x, f = threadIdx.x;
  float a2 = a2p[0];
  float zacc = 0.f;
  for (int p = 0; p < K_KEEP; ++p) {
    float v = gc2_raw[(size_t)g * 4096 + p * 64 + f] * scale2[f] + shift2[f];
    v = v >= 0.f ? v : a2 * v;
    out[((size_t)g * K_KEEP + p) * 64 + f] = v;
    zacc += v;
  }
  zacc *= (1.f / (float)K_KEEP);
  int b = g / TP_N, tp = g % TP_N;
  out[(size_t)NK_TOT * 64 + ((size_t)b * 64 + f) * TP_N + tp] = zacc;
}

// ================= host glue =================
extern "C" void kernel_launch(void* const* d_in, const int* in_sizes, int n_in,
                              void* d_out, int out_size, void* d_ws, size_t ws_size,
                              hipStream_t stream) {
  (void)in_sizes; (void)n_in; (void)out_size; (void)ws_size;
  const float* h   = (const float*)d_in[0];
  const int*   ei  = (const int*)d_in[1];
  const float* ew  = (const float*)d_in[2];
  const float* W1  = (const float*)d_in[3];
  const float* b1  = (const float*)d_in[4];
  const float* g1  = (const float*)d_in[5];
  const float* be1 = (const float*)d_in[6];
  const float* a1  = (const float*)d_in[7];
  const float* Wp  = (const float*)d_in[8];
  const float* bp  = (const float*)d_in[9];
  const float* W2  = (const float*)d_in[10];
  const float* b2  = (const float*)d_in[11];
  const float* g2  = (const float*)d_in[12];
  const float* be2 = (const float*)d_in[13];
  const float* a2  = (const float*)d_in[14];
  float* out = (float*)d_out;

  char* base = (char*)d_ws;
  size_t off = 0;
  auto carve = [&](size_t bytes) -> char* {
    char* q = base + off;
    off += (bytes + 255) & ~(size_t)255;
    return q;
  };
  _Float16* W1f   = (_Float16*)carve(128 * 128 * 2);                 // B-layout
  _Float16* W2f   = (_Float16*)carve(128 * 64 * 2);                  // B-layout
  _Float16* A1f   = (_Float16*)carve(64 * 64 * 2);                   // A-layout
  float*    Apf   = (float*)carve(64 * 64 * 4);                      // row-major
  _Float16* xk_f  = (_Float16*)carve((size_t)G_TOT * 8192 * 2);      // A-layout / graph
  _Float16* A2f_h = (_Float16*)carve((size_t)G_TOT * 4096 * 2);      // A-layout / graph
  float* psum1    = (float*)carve((size_t)G_TOT * 128 * 4);
  float* psumsq1  = (float*)carve((size_t)G_TOT * 128 * 4);
  float* scale1   = (float*)carve(128 * 4);
  float* shift1   = (float*)carve(128 * 4);
  float* psum2    = (float*)carve((size_t)G_TOT * 64 * 4);
  float* psumsq2  = (float*)carve((size_t)G_TOT * 64 * 4);
  float* scale2   = (float*)carve(64 * 4);
  float* shift2   = (float*)carve(64 * 4);
  float* gc1_raw  = (float*)carve((size_t)N_TOT * 128 * 4);
  float* gc2_raw  = gc1_raw;  // safe alias: gc1_raw fully consumed before k3_gc2 writes

  k0_prep<<<1, 256, 0, stream>>>(ei, ew, W1, W2, W1f, W2f, A1f, Apf);
  k1_gc1<<<G_TOT, 256, 0, stream>>>(h, W1f, A1f, b1, gc1_raw);
  k1b_stats<<<G_TOT, 128, 0, stream>>>(gc1_raw, psum1, psumsq1);
  k_bnred<<<1, 128, 0, stream>>>(psum1, psumsq1, g1, be1, scale1, shift1, 128, G_TOT,
                                 1.f / (float)N_TOT);
  k2_pool<<<G_TOT, 256, 0, stream>>>(gc1_raw, scale1, shift1, a1, Wp, bp, Apf, ei, ew,
                                     xk_f, A2f_h);
  k3_gc2<<<G_TOT, 256, 0, stream>>>(xk_f, W2f, A2f_h, b2, gc2_raw);
  k3b_stats<<<G_TOT, 64, 0, stream>>>(gc2_raw, psum2, psumsq2);
  k_bnred<<<1, 64, 0, stream>>>(psum2, psumsq2, g2, be2, scale2, shift2, 64, G_TOT,
                                1.f / (float)NK_TOT);
  k5_final<<<G_TOT, 64, 0, stream>>>(gc2_raw, scale2, shift2, a2, out);
}
